// LongTermMemoryMLP_64166811402842
// MI455X (gfx1250) — matrix operand order repaired and checked by reading, not compiled, not run on hardware
//
#include <hip/hip_runtime.h>

// ---------------------------------------------------------------------------
// Fused 3-layer batched MLP for gfx1250 (MI455X), bf16 WMMA with fp32 accum.
//   h0 = relu(q  @ W0^T + b0)   [B,S,1024], K=512
//   h1 = relu(h0 @ W1^T + b1)   [B,S,1024], K=1024
//   out =      h1 @ W2^T + b2   [B,S, 512], K=1024
// One workgroup = one (batch, 64-row tile of S). All intermediates in LDS.
// Weights are pre-converted fp32->bf16 once into d_ws (32 MiB) when it fits;
// otherwise an in-loop v_perm-based conversion fallback is used.
// ---------------------------------------------------------------------------

typedef __bf16        v16bf __attribute__((ext_vector_type(16)));
typedef float         v8f   __attribute__((ext_vector_type(8)));
typedef unsigned int  v4u   __attribute__((ext_vector_type(4)));
typedef unsigned int  v2u   __attribute__((ext_vector_type(2)));

#define NB    8
#define NS    4096
#define DIN   512
#define DH    1024
#define DOUT  512
#define TM    64              // rows of S per workgroup
#define QPITCH 520            // ushorts per q row  (512 + 8 pad -> 1040 B, 16B aligned)
#define HPITCH 1032           // ushorts per h row (1024 + 8 pad -> 2064 B, 16B aligned)

#define W0_ELTS ((size_t)NB * DH * DIN)    // 4,194,304
#define W1_ELTS ((size_t)NB * DH * DH)     // 8,388,608
#define W2_ELTS ((size_t)NB * DOUT * DH)   // 4,194,304
#define WS_NEEDED ((W0_ELTS + W1_ELTS + W2_ELTS) * 2)   // 33,554,432 B

// fp32 -> bf16 bits, round-half-up (1 add + 1 shift; ties differ from RNE only)
__device__ __forceinline__ unsigned short f2bf(float f) {
  union { float f; unsigned int u; } v; v.f = f;
  return (unsigned short)((v.u + 0x8000u) >> 16);
}

// two fp32 -> packed bf16x2: 2 adds + 1 v_perm_b32 (3 VALU for 2 elements)
__device__ __forceinline__ unsigned int f2bf2(float lo, float hi) {
  union { float f; unsigned int u; } a, b; a.f = lo; b.f = hi;
  const unsigned int x = a.u + 0x8000u;
  const unsigned int y = b.u + 0x8000u;
  // dst = { y[31:16], x[31:16] } : S0=y supplies bytes 4..7, S1=x bytes 0..3
  return __builtin_amdgcn_perm(y, x, 0x07060302u);
}

// A fragment: two contiguous 16B chunks of bf16 (contiguous A layout:
// lane L holds row L&15; K = kh..kh+15, kh=(L>=16)*16)
__device__ __forceinline__ v16bf load_a16(const unsigned short* p) {
  union { v16bf bf; v4u u[2]; } f;
  f.u[0] = *(const v4u*)(p);
  f.u[1] = *(const v4u*)(p + 8);
  return f.bf;
}

// B fragment from pre-converted bf16 weights: 32 contiguous bytes, no VALU
// (contiguous B layout, same k order as A: lane L holds column L&15, contiguous K from (L>=16)*16)
__device__ __forceinline__ v16bf load_b16_bf(const unsigned short* p) {
  union { v16bf bf; v4u u[2]; } f;
  f.u[0] = ((const v4u*)p)[0];
  f.u[1] = ((const v4u*)p)[1];
  return f.bf;
}

// B fragment from fp32 weights (fallback): 64B loads + 8 perm-packs
__device__ __forceinline__ v16bf load_b16_f32(const float* p) {
  float4 x0 = ((const float4*)p)[0];
  float4 x1 = ((const float4*)p)[1];
  float4 x2 = ((const float4*)p)[2];
  float4 x3 = ((const float4*)p)[3];
  union { v16bf bf; unsigned int u[8]; } f;
  f.u[0] = f2bf2(x0.x, x0.y);  f.u[1] = f2bf2(x0.z, x0.w);
  f.u[2] = f2bf2(x1.x, x1.y);  f.u[3] = f2bf2(x1.z, x1.w);
  f.u[4] = f2bf2(x2.x, x2.y);  f.u[5] = f2bf2(x2.z, x2.w);
  f.u[6] = f2bf2(x3.x, x3.y);  f.u[7] = f2bf2(x3.z, x3.w);
  return f.bf;
}

// One layer: this wave computes a 64 x (NTILES*16) slab of the output.
// K is also the row stride of W ([out, K] row-major).
template<int K, int NTILES, bool RELU, bool TOLDS, bool BF16W>
__device__ __forceinline__ void layer_gemm(
    const unsigned short* __restrict__ Alds, int apitch,
    const float* __restrict__ Wf,            // fp32 weights (fallback path)
    const unsigned short* __restrict__ Wb,   // bf16 weights (fast path)
    const float* __restrict__ bias,
    unsigned short* __restrict__ Hlds,       // used when TOLDS
    float* __restrict__ Gout,                // used when !TOLDS (pitch DOUT)
    int wave, int lane)
{
  const int r  = lane & 15;
  const int hi = lane >> 4;
  const int kh = hi << 4;   // A-fragment K sub-offset
  const int kb = hi << 4;   // B-fragment K sub-offset
  const int r0 = hi << 3;   // C/D row sub-offset

  for (int nt = 0; nt < NTILES; ++nt) {
    const int n = (wave * NTILES + nt) * 16;
    v8f acc[4] = {};
    if constexpr (BF16W) {
      const unsigned short* wrow = Wb + (size_t)(n + r) * K;
#pragma unroll 2
      for (int k0 = 0; k0 < K; k0 += 32) {
        const v16bf bf = load_b16_bf(wrow + k0 + kb);
#pragma unroll
        for (int mt = 0; mt < 4; ++mt) {
          const unsigned short* ap = Alds + (size_t)(mt * 16 + r) * apitch + k0 + kh;
          acc[mt] = __builtin_amdgcn_wmma_f32_16x16x32_bf16(
              false, load_a16(ap), false, bf, (short)0, acc[mt], false, false);
        }
      }
    } else {
      const float* wrow = Wf + (size_t)(n + r) * K;
#pragma unroll 2
      for (int k0 = 0; k0 < K; k0 += 32) {
        const v16bf bf = load_b16_f32(wrow + k0 + kb);
#pragma unroll
        for (int mt = 0; mt < 4; ++mt) {
          const unsigned short* ap = Alds + (size_t)(mt * 16 + r) * apitch + k0 + kh;
          acc[mt] = __builtin_amdgcn_wmma_f32_16x16x32_bf16(
              false, load_a16(ap), false, bf, (short)0, acc[mt], false, false);
        }
      }
    }
    const float bv = bias[n + r];
#pragma unroll
    for (int mt = 0; mt < 4; ++mt) {
#pragma unroll
      for (int j = 0; j < 8; ++j) {
        float v = acc[mt][j] + bv;
        if (RELU) v = fmaxf(v, 0.0f);
        const int row = mt * 16 + r0 + j;
        if constexpr (TOLDS) Hlds[(size_t)row * HPITCH + n + r] = f2bf(v);
        else                 Gout[(size_t)row * DOUT   + n + r] = v;
      }
    }
  }
}

// One-shot fp32 -> bf16 weight conversion: 8 elements / thread, 16B stores.
__global__ void __launch_bounds__(256, 4)
convert_f32_bf16(const float* __restrict__ src, unsigned short* __restrict__ dst) {
  const size_t i = ((size_t)blockIdx.x * 256 + threadIdx.x) * 8;
  const float4 a = ((const float4*)(src + i))[0];
  const float4 b = ((const float4*)(src + i))[1];
  v4u o;
  o.x = f2bf2(a.x, a.y);
  o.y = f2bf2(a.z, a.w);
  o.z = f2bf2(b.x, b.y);
  o.w = f2bf2(b.z, b.w);
  *(v4u*)(dst + i) = o;
}

template<bool BF16W>
__global__ void __launch_bounds__(256, 1)
mlp3_fused_bf16_wmma(const float* __restrict__ q,
                     const float* __restrict__ W0f, const unsigned short* __restrict__ W0b,
                     const float* __restrict__ b0,
                     const float* __restrict__ W1f, const unsigned short* __restrict__ W1b,
                     const float* __restrict__ b1,
                     const float* __restrict__ W2f, const unsigned short* __restrict__ W2b,
                     const float* __restrict__ b2,
                     float* __restrict__ out)
{
  // Region X holds q (first 65 KB) during layer 0, then h1 during layers 1/2.
  __shared__ unsigned short ldsX [TM * HPITCH];   // 129 KB
  __shared__ unsigned short ldsH0[TM * HPITCH];   // 129 KB

  const int b     = blockIdx.y;
  const int sBase = blockIdx.x * TM;
  const int tid   = threadIdx.x;
  const int wave  = tid >> 5;
  const int lane  = tid & 31;

  // ---- stage query tile into LDS as bf16 (perm-pack + 8B ds stores) ------
  const float* qg = q + ((size_t)b * NS + sBase) * DIN;
  for (int c = tid; c < TM * (DIN / 4); c += 256) {
    const int row  = c >> 7;           // DIN/4 == 128
    const int col4 = (c & 127) << 2;
    const float4 f = *(const float4*)(qg + (size_t)row * DIN + col4);
    v2u pk;
    pk.x = f2bf2(f.x, f.y);
    pk.y = f2bf2(f.z, f.w);
    *(v2u*)(ldsX + (size_t)row * QPITCH + col4) = pk;
  }
  __syncthreads();

  // ---- layer 0: q(LDS) x W0 -> h0(LDS), relu, K=512 ----------------------
  layer_gemm<DIN, 8, true, true, BF16W>(ldsX, QPITCH,
      W0f + (BF16W ? 0 : (size_t)b * DH * DIN),
      W0b + (BF16W ? (size_t)b * DH * DIN : 0),
      b0 + (size_t)b * DH, ldsH0, nullptr, wave, lane);
  __syncthreads();

  // ---- layer 1: h0(LDS) x W1 -> h1(LDS, overlays q), relu, K=1024 --------
  layer_gemm<DH, 8, true, true, BF16W>(ldsH0, HPITCH,
      W1f + (BF16W ? 0 : (size_t)b * DH * DH),
      W1b + (BF16W ? (size_t)b * DH * DH : 0),
      b1 + (size_t)b * DH, ldsX, nullptr, wave, lane);
  __syncthreads();

  // ---- layer 2: h1(LDS) x W2 -> out(global), +b2, K=1024 -----------------
  layer_gemm<DH, 4, false, false, BF16W>(ldsX, HPITCH,
      W2f + (BF16W ? 0 : (size_t)b * DOUT * DH),
      W2b + (BF16W ? (size_t)b * DOUT * DH : 0),
      b2 + (size_t)b * DOUT, nullptr,
      out + ((size_t)b * NS + sBase) * DOUT, wave, lane);
}

extern "C" void kernel_launch(void* const* d_in, const int* in_sizes, int n_in,
                              void* d_out, int out_size, void* d_ws, size_t ws_size,
                              hipStream_t stream) {
  (void)in_sizes; (void)n_in; (void)out_size;
  const float* q  = (const float*)d_in[0];
  const float* W0 = (const float*)d_in[1];
  const float* b0 = (const float*)d_in[2];
  const float* W1 = (const float*)d_in[3];
  const float* b1 = (const float*)d_in[4];
  const float* W2 = (const float*)d_in[5];
  const float* b2 = (const float*)d_in[6];
  float* out = (float*)d_out;

  dim3 grid(NS / TM, NB);   // 64 s-tiles x 8 batches = 512 workgroups
  dim3 block(256);          // 8 wave32

  if (d_ws != nullptr && ws_size >= (size_t)WS_NEEDED) {
    // Fast path: one-shot weight conversion into d_ws, then pure-bf16 GEMM.
    unsigned short* W0b = (unsigned short*)d_ws;
    unsigned short* W1b = W0b + W0_ELTS;
    unsigned short* W2b = W1b + W1_ELTS;
    convert_f32_bf16<<<dim3((unsigned)(W0_ELTS / (256 * 8))), block, 0, stream>>>(W0, W0b);
    convert_f32_bf16<<<dim3((unsigned)(W1_ELTS / (256 * 8))), block, 0, stream>>>(W1, W1b);
    convert_f32_bf16<<<dim3((unsigned)(W2_ELTS / (256 * 8))), block, 0, stream>>>(W2, W2b);
    mlp3_fused_bf16_wmma<true><<<grid, block, 0, stream>>>(
        q, W0, W0b, b0, W1, W1b, b1, W2, W2b, b2, out);
  } else {
    // Fallback: convert weights in-loop with v_perm packing.
    mlp3_fused_bf16_wmma<false><<<grid, block, 0, stream>>>(
        q, W0, (const unsigned short*)nullptr, b0,
        W1, (const unsigned short*)nullptr, b1,
        W2, (const unsigned short*)nullptr, b2, out);
  }
}
